// LPF_18640158065294
// MI455X (gfx1250) — matrix operand order repaired and checked by reading, not compiled, not run on hardware
//
#include <hip/hip_runtime.h>

// ---------------------------------------------------------------------------
// MI455X (gfx1250, wave32) implementation.
// Matrix work (Gram matrix for kNN, aggregator linears) runs on
// v_wmma_f32_16x16x32_bf16 (K=32 == channel dim, one instruction per tile).
// The kNN kernel stages the per-batch feature tile (64KB) into LDS with the
// Tensor Data Mover (tensor_load_to_lds + s_wait_tensorcnt) so all 64 column
// tiles are fed from LDS instead of re-reading L2.
// Everything is L2-resident (~49MB workspace << 192MB L2).
// ---------------------------------------------------------------------------

typedef __attribute__((ext_vector_type(16))) __bf16 v16bf;
typedef __attribute__((ext_vector_type(8)))  float  v8f;
typedef __attribute__((ext_vector_type(4)))  unsigned int u32x4;
typedef __attribute__((ext_vector_type(8)))  int    i32x8;
typedef __attribute__((ext_vector_type(4)))  int    i32x4;

union BfFrag {
    v16bf v;
    unsigned short u[16];
    uint4 q[2];
};

__device__ __forceinline__ unsigned short f2bf(float f) {
    unsigned int u = __float_as_uint(f);
    return (unsigned short)((u + 0x7FFFu + ((u >> 16) & 1u)) >> 16); // RNE
}
__device__ __forceinline__ float bf2f(unsigned short h) {
    return __uint_as_float(((unsigned int)h) << 16);
}
__device__ __forceinline__ float leaky(float v) {
    return v >= 0.0f ? v : 0.05f * v;
}

// Problem constants
#define BB   64
#define CH   32
#define HW   1024
#define NROW 65536L   // BB*HW

// ---------------------------------------------------------------------------
// 1) 5x5 SAME conv (cross-correlation), NCHW. One block per (b, oc).
// ---------------------------------------------------------------------------
__global__ __launch_bounds__(256) void conv5x5_kernel(
    const float* __restrict__ x, const float* __restrict__ cw,
    const float* __restrict__ cb, float* __restrict__ y)
{
    __shared__ float tile[36 * 36];
    const int tid = threadIdx.x;
    const int b  = blockIdx.x >> 5;
    const int oc = blockIdx.x & 31;
    float acc[4] = {0.f, 0.f, 0.f, 0.f};

    for (int ic = 0; ic < CH; ++ic) {
        for (int t = tid; t < 36 * 36; t += 256) {
            int py = t / 36 - 2, px = t % 36 - 2;
            float v = 0.f;
            if (py >= 0 && py < 32 && px >= 0 && px < 32)
                v = x[((long)(b * CH + ic) << 10) + py * 32 + px];
            tile[t] = v;
        }
        if (ic + 1 < CH)  // gfx1250: lowers to global_prefetch_b8
            __builtin_prefetch(&x[((long)(b * CH + ic + 1) << 10) + (tid << 2)], 0, 1);
        float wk[25];
        const float* wp = cw + (oc * CH + ic) * 25;
        #pragma unroll
        for (int k = 0; k < 25; ++k) wk[k] = wp[k];
        __syncthreads();
        #pragma unroll
        for (int p = 0; p < 4; ++p) {
            int pix = tid + p * 256;
            int h = pix >> 5, wc = pix & 31;
            float s = acc[p];
            #pragma unroll
            for (int dy = 0; dy < 5; ++dy)
                #pragma unroll
                for (int dx = 0; dx < 5; ++dx)
                    s = fmaf(tile[(h + dy) * 36 + wc + dx], wk[dy * 5 + dx], s);
            acc[p] = s;
        }
        __syncthreads();
    }
    #pragma unroll
    for (int p = 0; p < 4; ++p)
        y[((long)(b * CH + oc) << 10) + tid + p * 256] = acc[p] + cb[oc];
}

// ---------------------------------------------------------------------------
// 2) Batch-norm statistics (training mode, biased var). One block per channel.
//    nchw=1: src is (B,C,HW).  nchw=0: src is (N,C).
//    stat[c]=mean, stat[32+c]=rsqrt(var+eps)
// ---------------------------------------------------------------------------
__global__ __launch_bounds__(256) void bn_stats_kernel(
    const float* __restrict__ src, float* __restrict__ stat, int nchw)
{
    __shared__ float s1s[256], s2s[256];
    const int c = blockIdx.x, tid = threadIdx.x;
    float s1 = 0.f, s2 = 0.f;
    for (long t = tid; t < NROW; t += 256) {
        float v;
        if (nchw) {
            long b = t >> 10, hw = t & 1023;
            v = src[((b * CH + c) << 10) + hw];
        } else {
            v = src[t * CH + c];
        }
        s1 += v; s2 += v * v;
    }
    s1s[tid] = s1; s2s[tid] = s2;
    __syncthreads();
    for (int off = 128; off > 0; off >>= 1) {
        if (tid < off) { s1s[tid] += s1s[tid + off]; s2s[tid] += s2s[tid + off]; }
        __syncthreads();
    }
    if (tid == 0) {
        float m   = s1s[0] * (1.0f / 65536.0f);
        float var = s2s[0] * (1.0f / 65536.0f) - m * m;
        stat[c]      = m;
        stat[CH + c] = rsqrtf(var + 1e-5f);
    }
}

// ---------------------------------------------------------------------------
// 3) First BN apply + leaky, NCHW -> (N,C) fp32 + bf16, plus per-row ||f||^2
//    (squared norm of the bf16-rounded values, for WMMA-consistent distances)
// ---------------------------------------------------------------------------
__global__ __launch_bounds__(256) void bn_apply_first_kernel(
    const float* __restrict__ y, const float* __restrict__ stat,
    const float* __restrict__ gam, const float* __restrict__ bet,
    float* __restrict__ f, unsigned short* __restrict__ fb,
    float* __restrict__ sq)
{
    const long n = (long)blockIdx.x * 256 + threadIdx.x;  // 0..65535
    const long b = n >> 10, hw = n & 1023;
    float s = 0.f;
    #pragma unroll
    for (int c = 0; c < CH; ++c) {
        float v = (y[((b * CH + c) << 10) + hw] - stat[c]) * stat[CH + c] * gam[c] + bet[c];
        v = leaky(v);
        f[n * CH + c] = v;
        unsigned short h = f2bf(v);
        fb[n * CH + c] = h;
        float vb = bf2f(h);
        s += vb * vb;
    }
    sq[n] = s;
}

// ---------------------------------------------------------------------------
// 4) Per-batch Gram matrix via WMMA + fused top-8 (smallest distance) select.
//    Block = 128 threads = 4 waves; wave handles a 16-row strip; 64 column
//    tiles -> 64 v_wmma_f32_16x16x32_bf16 per wave.
//    The batch feature tile (1024x32 bf16 = 64KB) is staged into LDS by the
//    Tensor Data Mover (D# per ISA ch.8: 1-D tile of 8192 x 8-byte elements),
//    then all fragments are ds_load_b128 from LDS.
//    Fragment layouts per CDNA5 ISA 7.12.2:
//      A (16x32 bf16): lane l (row=l&15), halves e: K = e + 8*(e>=8) + 8*hi
//      B (32x16 bf16): lane l (col=l&15), halves e: K = e + 8*(e>=8) + 8*hi
//      C (16x16 f32) : vgpr v -> (M = v + 8*hi, N = l&15)
// ---------------------------------------------------------------------------
__global__ __launch_bounds__(128) void dist_topk_kernel(
    const unsigned short* __restrict__ fb, const float* __restrict__ sq,
    int* __restrict__ top8)
{
    __shared__ unsigned short ftile[HW * CH];   // 64KB: whole batch f (bf16)
    __shared__ float dt[4][16][16];
    __shared__ float mv[4][16][16];
    __shared__ int   mi[4][16][16];

    const int tid  = threadIdx.x;
    const int w    = tid >> 5;
    const int lane = tid & 31;
    const int hi   = lane >> 4;
    const int l15  = lane & 15;
    const int b      = blockIdx.x >> 4;
    const int tile64 = blockIdx.x & 15;
    const int r0     = tile64 * 64 + w * 16;      // row base within batch
    const long nbase = (long)b * HW;

    // ---- TDM: stage this batch's 64KB bf16 feature tile into LDS ----------
    if (tid < 32) {
        unsigned long long ga = (unsigned long long)(uintptr_t)(fb + nbase * CH);
        unsigned ldsoff = (unsigned)(uintptr_t)&ftile[0];
        // D# group0: count=1, lds_addr, global_addr[56:0], type=2
        u32x4 g0 = {1u, ldsoff,
                    (unsigned)(ga & 0xFFFFFFFFu),
                    (unsigned)((ga >> 32) & 0x01FFFFFFu) | 0x80000000u};
        // D# group1: data_size=8B (code 3); tensor_dim0=8192, tensor_dim1=1,
        // tile_dim0=8192, tile_dim1=1, tensor_dim0_stride=8192; no multicast,
        // no padding, no iteration, no atomic barrier.
        i32x8 g1 = {0x00030000, 0x20000000, 0x00010000, 0x20000000,
                    1, 8192, 0, 0};
        i32x4 gz = {0, 0, 0, 0};
#if __has_include(<hip/amd_detail/amd_gfx1250_TDM.h>)
        // amdgpu-toolchain (therock headers present): 6-arg builtin
        i32x8 gz8 = {0, 0, 0, 0, 0, 0, 0, 0};
        __builtin_amdgcn_tensor_load_to_lds(g0, g1, gz, gz, gz8, 0);
#else
        // ROCm 7.2: 5-arg builtin
        __builtin_amdgcn_tensor_load_to_lds(g0, g1, gz, gz, 0);
#endif
        __builtin_amdgcn_s_wait_tensorcnt(0);
    }
    __syncthreads();

    // loop-invariant A fragment: rows r0..r0+15, K=0..31 (from LDS)
    BfFrag afrag;
    {
        const uint4* p = (const uint4*)(ftile + (r0 + l15) * CH);
        afrag.q[0] = p[hi];
        afrag.q[1] = p[2 + hi];
    }
    float sqi[8];
    #pragma unroll
    for (int v = 0; v < 8; ++v) sqi[v] = sq[nbase + r0 + v + hi * 8];

    // private sorted (ascending) top-8
    float tv[8]; int ti[8];
    #pragma unroll
    for (int k = 0; k < 8; ++k) { tv[k] = 3.0e38f; ti[k] = 0; }
    const int selrow = lane >> 1;   // 2 lanes per row
    const int selg   = lane & 1;    // column half

    for (int j0 = 0; j0 < HW; j0 += 16) {
        BfFrag bfrag;
        const uint4* p = (const uint4*)(ftile + (j0 + l15) * CH);
        bfrag.q[0] = p[hi];
        bfrag.q[1] = p[2 + hi];

        v8f c = {};
        c = __builtin_amdgcn_wmma_f32_16x16x32_bf16(
                false, afrag.v, false, bfrag.v, (short)0, c, false, false);

        float sqj = sq[nbase + j0 + l15];
        #pragma unroll
        for (int v = 0; v < 8; ++v)
            dt[w][v + hi * 8][l15] = sqi[v] - 2.0f * c[v] + sqj;
        __syncthreads();

        #pragma unroll
        for (int t = 0; t < 8; ++t) {
            float d = dt[w][selrow][selg * 8 + t];
            int   j = j0 + selg * 8 + t;
            if (d < tv[7]) {                 // sorted insert (one bubble pass)
                tv[7] = d; ti[7] = j;
                #pragma unroll
                for (int s = 7; s > 0; --s) {
                    if (tv[s] < tv[s - 1]) {
                        float ft = tv[s]; tv[s] = tv[s - 1]; tv[s - 1] = ft;
                        int   it = ti[s]; ti[s] = ti[s - 1]; ti[s - 1] = it;
                    }
                }
            }
        }
        __syncthreads();
    }

    // merge the two lanes' sorted lists per row, emit 8 smallest indices
    #pragma unroll
    for (int k = 0; k < 8; ++k) {
        mv[w][selrow][selg * 8 + k] = tv[k];
        mi[w][selrow][selg * 8 + k] = ti[k];
    }
    __syncthreads();
    if (selg == 0) {
        int pa = 0, pb = 0;
        const long n = nbase + r0 + selrow;
        #pragma unroll
        for (int k = 0; k < 8; ++k) {
            float va = (pa < 8) ? mv[w][selrow][pa]     : 3.0e38f;
            float vb = (pb < 8) ? mv[w][selrow][8 + pb] : 3.0e38f;
            if (va <= vb) { top8[n * 8 + k] = mi[w][selrow][pa];     ++pa; }
            else          { top8[n * 8 + k] = mi[w][selrow][8 + pb]; ++pb; }
        }
    }
}

// ---------------------------------------------------------------------------
// 5) gmean[n,c] = f[n,c] - mean_k f[nbr_k, c]   (indices are within-batch)
// ---------------------------------------------------------------------------
__global__ __launch_bounds__(256) void gmean_kernel(
    const float* __restrict__ f, const int* __restrict__ top8,
    float* __restrict__ g0, unsigned short* __restrict__ gb)
{
    const long t = (long)blockIdx.x * 256 + threadIdx.x;  // < NROW*CH
    const long n = t >> 5;
    const int  c = (int)(t & 31);
    const long bbase = (n >> 10) << 10;
    float s = 0.f;
    #pragma unroll
    for (int k = 0; k < 8; ++k) {
        int j = top8[n * 8 + k];
        s += f[(bbase + j) * CH + c];
    }
    float v = f[t] - s * 0.125f;
    g0[t] = v;
    gb[t] = f2bf(v);
}

// ---------------------------------------------------------------------------
// 6) Aggregator GEMM: z[n,o] = sum_c g[n,c]*dw[o,c] + sum_c f[n,c]*sw[o,c]
//                              + db[o] + sb[o]
//    Wave = one 16-row tile, two 16-col output tiles, K=32 in one shot:
//    4 v_wmma per wave. Weights converted to bf16 in-register (loop-invariant).
// ---------------------------------------------------------------------------
__global__ __launch_bounds__(256) void agg_gemm_kernel(
    const unsigned short* __restrict__ fin_b, const unsigned short* __restrict__ gb,
    const float* __restrict__ dw, const float* __restrict__ db,
    const float* __restrict__ sw, const float* __restrict__ sb,
    float* __restrict__ z)
{
    const int tid  = threadIdx.x;
    const int w    = tid >> 5;
    const int lane = tid & 31;
    const int hi   = lane >> 4;
    const int l15  = lane & 15;
    const long m0  = ((long)blockIdx.x * 8 + w) * 16;

    BfFrag bdw[2], bsw[2];
    float bias[2];
    #pragma unroll
    for (int ot = 0; ot < 2; ++ot) {
        int o = ot * 16 + l15;
        const float* pd = dw + o * CH + hi * 16;
        const float* ps = sw + o * CH + hi * 16;
        #pragma unroll
        for (int e = 0; e < 16; ++e) {
            bdw[ot].u[e] = f2bf(pd[e]);
            bsw[ot].u[e] = f2bf(ps[e]);
        }
        bias[ot] = db[o] + sb[o];
    }

    BfFrag ag, af;
    {
        const uint4* pg = (const uint4*)(gb + (m0 + l15) * CH);
        ag.q[0] = pg[2 * hi]; ag.q[1] = pg[2 * hi + 1];
        const uint4* pf = (const uint4*)(fin_b + (m0 + l15) * CH);
        af.q[0] = pf[2 * hi]; af.q[1] = pf[2 * hi + 1];
    }

    #pragma unroll
    for (int ot = 0; ot < 2; ++ot) {
        v8f c = {};
        c = __builtin_amdgcn_wmma_f32_16x16x32_bf16(
                false, ag.v, false, bdw[ot].v, (short)0, c, false, false);
        c = __builtin_amdgcn_wmma_f32_16x16x32_bf16(
                false, af.v, false, bsw[ot].v, (short)0, c, false, false);
        #pragma unroll
        for (int v = 0; v < 8; ++v) {
            long n = m0 + v + hi * 8;
            z[n * CH + ot * 16 + l15] = c[v] + bias[ot];
        }
    }
}

// ---------------------------------------------------------------------------
// 7) BN apply + leaky on (N,C), write fp32 + bf16
// ---------------------------------------------------------------------------
__global__ __launch_bounds__(256) void bn_apply_nc_kernel(
    const float* __restrict__ z, const float* __restrict__ stat,
    const float* __restrict__ gam, const float* __restrict__ bet,
    float* __restrict__ f, unsigned short* __restrict__ fb)
{
    const long t = (long)blockIdx.x * 256 + threadIdx.x;
    const int  c = (int)(t & 31);
    float v = (z[t] - stat[c]) * stat[CH + c] * gam[c] + bet[c];
    v = leaky(v);
    f[t]  = v;
    fb[t] = f2bf(v);
}

// ---------------------------------------------------------------------------
// 8) Residual: out (NCHW) = x + f ((N,C))
// ---------------------------------------------------------------------------
__global__ __launch_bounds__(256) void final_add_kernel(
    const float* __restrict__ x, const float* __restrict__ f,
    float* __restrict__ out)
{
    const long t = (long)blockIdx.x * 256 + threadIdx.x;   // < B*C*HW
    const long b = t >> 15;
    const long r = t & 32767;
    const int  c  = (int)(r >> 10);
    const int  hw = (int)(r & 1023);
    out[t] = x[t] + f[((b << 10) + hw) * CH + c];
}

// ---------------------------------------------------------------------------
// Workspace layout (float slots); z aliases y (y dead after first BN apply).
// ---------------------------------------------------------------------------
static constexpr long M2       = NROW * CH;          // 2,097,152 floats
static constexpr long OFF_Y    = 0;                  // y / z (aliased)
static constexpr long OFF_F0   = 1 * M2;
static constexpr long OFF_G0   = 2 * M2;
static constexpr long OFF_F1   = 3 * M2;
static constexpr long OFF_FBA  = 4 * M2;             // bf16 (M2 halves)
static constexpr long OFF_FBB  = 4 * M2 + M2 / 2;
static constexpr long OFF_GB   = 5 * M2;
static constexpr long OFF_SQ   = 5 * M2 + M2 / 2;
static constexpr long OFF_TOP8 = OFF_SQ + NROW;      // int slots
static constexpr long OFF_STAT = OFF_TOP8 + NROW * 8;

extern "C" void kernel_launch(void* const* d_in, const int* in_sizes, int n_in,
                              void* d_out, int out_size, void* d_ws, size_t ws_size,
                              hipStream_t stream)
{
    const float* x    = (const float*)d_in[0];
    const float* cw   = (const float*)d_in[1];
    const float* cb   = (const float*)d_in[2];
    const float* bn0g = (const float*)d_in[3];
    const float* bn0b = (const float*)d_in[4];
    const float* dwp[3] = {(const float*)d_in[5],  (const float*)d_in[11], (const float*)d_in[17]};
    const float* dbp[3] = {(const float*)d_in[6],  (const float*)d_in[12], (const float*)d_in[18]};
    const float* swp[3] = {(const float*)d_in[7],  (const float*)d_in[13], (const float*)d_in[19]};
    const float* sbp[3] = {(const float*)d_in[8],  (const float*)d_in[14], (const float*)d_in[20]};
    const float* bgp[3] = {(const float*)d_in[9],  (const float*)d_in[15], (const float*)d_in[21]};
    const float* bbp[3] = {(const float*)d_in[10], (const float*)d_in[16], (const float*)d_in[22]};

    float* ws = (float*)d_ws;
    float* y  = ws + OFF_Y;
    float* z  = ws + OFF_Y;       // alias
    float* f0 = ws + OFF_F0;
    float* g0 = ws + OFF_G0;
    float* f1 = ws + OFF_F1;
    unsigned short* fbA = (unsigned short*)(ws + OFF_FBA);
    unsigned short* fbB = (unsigned short*)(ws + OFF_FBB);
    unsigned short* gb  = (unsigned short*)(ws + OFF_GB);
    float* sq   = ws + OFF_SQ;
    int*   top8 = (int*)(ws + OFF_TOP8);
    float* stat = ws + OFF_STAT;
    float* out  = (float*)d_out;

    conv5x5_kernel<<<BB * CH, 256, 0, stream>>>(x, cw, cb, y);
    bn_stats_kernel<<<CH, 256, 0, stream>>>(y, stat, 1);
    bn_apply_first_kernel<<<NROW / 256, 256, 0, stream>>>(y, stat, bn0g, bn0b, f0, fbA, sq);
    dist_topk_kernel<<<BB * 16, 128, 0, stream>>>(fbA, sq, top8);
    gmean_kernel<<<(NROW * CH) / 256, 256, 0, stream>>>(f0, top8, g0, gb);

    float* fin = f0;  unsigned short* finb = fbA;
    float* fout = f1; unsigned short* foutb = fbB;
    for (int L = 0; L < 3; ++L) {
        agg_gemm_kernel<<<NROW / 16 / 8, 256, 0, stream>>>(finb, gb, dwp[L], dbp[L],
                                                           swp[L], sbp[L], z);
        bn_stats_kernel<<<CH, 256, 0, stream>>>(z, stat, 0);
        bn_apply_nc_kernel<<<(NROW * CH) / 256, 256, 0, stream>>>(z, stat, bgp[L], bbp[L],
                                                                  fout, foutb);
        float* tf = fin; fin = fout; fout = tf;
        unsigned short* tb = finb; finb = foutb; foutb = tb;
    }
    final_add_kernel<<<(NROW * CH) / 256, 256, 0, stream>>>(x, fin, out);
}
